// MultiHeadedAttention_27066883900001
// MI455X (gfx1250) — matrix operand order repaired and checked by reading, not compiled, not run on hardware
//
#include <hip/hip_runtime.h>
#include <hip/hip_bf16.h>

// ---------------------------------------------------------------------------
// Fused MHA for MI455X (gfx1250, wave32, WMMA bf16 -> f32 accumulate)
// B=4, S=1024, D=1024, H=16, HD=64
// d_out = [ out (B*S*D fp32) | att (B*H*S*S fp32) ]
// ---------------------------------------------------------------------------

typedef __attribute__((ext_vector_type(16))) __bf16 v16bf;
typedef __attribute__((ext_vector_type(8)))  __bf16 v8bf;
typedef __attribute__((ext_vector_type(8)))  float  v8f;
typedef __attribute__((ext_vector_type(4)))  float  v4f;
typedef __attribute__((ext_vector_type(2)))  float  v2f;

#define WMMA_BF16(a, b, c) \
  __builtin_amdgcn_wmma_f32_16x16x32_bf16(false, (a), false, (b), (short)0, (c), false, false)

// Build a v16bf fragment from two 16-byte chunks (global or LDS).
__device__ __forceinline__ v16bf frag2(const __bf16* p0, const __bf16* p1) {
  v8bf lo = *(const v8bf*)p0;
  v8bf hi = *(const v8bf*)p1;
  v16bf f;
#pragma unroll
  for (int i = 0; i < 8; ++i) { f[i] = lo[i]; f[i + 8] = hi[i]; }
  return f;
}

// A-matrix (16x32 bf16): lane = M, halves K = {hi8+0..7, 16+hi8+0..7}.
__device__ __forceinline__ v16bf fragA(const __bf16* rowBase, int hi8) {
  const __bf16* p = rowBase + hi8;
  return frag2(p, p + 16);
}
// B-matrix (32x16 bf16): lane = N, halves K = {hi8+0..7, 16+hi8+0..7}.
__device__ __forceinline__ v16bf fragB(const __bf16* colBase, int hi16) {
  const __bf16* p = colBase + (hi16 >> 1);
  return frag2(p, p + 16);
}

__device__ __forceinline__ unsigned pack_bf16x2(float a, float b) {
  union { __bf16 h; unsigned short u; } ua, ub;
  ua.h = (__bf16)a; ub.h = (__bf16)b;
  return (unsigned)ua.u | ((unsigned)ub.u << 16);
}

// ------------------------------ converters ---------------------------------

__global__ void cvt_f32_to_bf16_v4(const float* __restrict__ src,
                                   __bf16* __restrict__ dst, int n4) {
  int i = blockIdx.x * 256 + threadIdx.x;
  if (i >= n4) return;
  v4f v = ((const v4f*)src)[i];
  uint2 p;
  p.x = pack_bf16x2(v.x, v.y);
  p.y = pack_bf16x2(v.z, v.w);
  ((uint2*)dst)[i] = p;
}

// Wt[n*K + k] = (bf16) W[k*N + n]   (W is [K,N] fp32 row-major)
__global__ void transpose_to_bf16(const float* __restrict__ W,
                                  __bf16* __restrict__ Wt, int K, int N) {
  int idx = blockIdx.x * 256 + threadIdx.x;
  if (idx >= K * N) return;
  int k = idx / N, n = idx - k * N;
  Wt[(size_t)n * K + k] = (__bf16)W[idx];
}

// ------------------------------ QKV GEMM -----------------------------------
// C[4096,3072] = Xb @ WqkvT^T + bqkv, scattered into Q/K/Vt (bf16).
// 256 thr = 8 waves (2M x 4N); wave tile 32(M) x 64(N); K step 32 -> 8 WMMA/iter.
__global__ __launch_bounds__(256) void qkv_gemm(
    const __bf16* __restrict__ Xb,     // [4096,1024]
    const __bf16* __restrict__ WqkvT,  // [3072,1024]
    const float*  __restrict__ bqkv,   // [3072]
    __bf16* __restrict__ Qb,           // [64,1024,64]  (bh, s, hd)
    __bf16* __restrict__ Kb,           // [64,1024,64]
    __bf16* __restrict__ Vt) {         // [64,64,1024]  (bh, hd, s)
  const int KDIM = 1024;
  const int wave = threadIdx.x >> 5, lane = threadIdx.x & 31;
  const int lo16 = lane & 15;
  const int hi8  = (lane >> 4) * 8;
  const int hi16 = (lane >> 4) * 16;
  const int m0 = blockIdx.y * 64  + (wave >> 2) * 32;
  const int n0 = blockIdx.x * 256 + (wave & 3) * 64;

  v8f acc[2][4] = {};
  for (int k0 = 0; k0 < KDIM; k0 += 32) {
    v16bf a[2], b[4];
#pragma unroll
    for (int mt = 0; mt < 2; ++mt)
      a[mt] = fragA(Xb + (size_t)(m0 + mt * 16 + lo16) * KDIM + k0, hi8);
#pragma unroll
    for (int nt = 0; nt < 4; ++nt)
      b[nt] = fragB(WqkvT + (size_t)(n0 + nt * 16 + lo16) * KDIM + k0, hi16);
#pragma unroll
    for (int mt = 0; mt < 2; ++mt)
#pragma unroll
      for (int nt = 0; nt < 4; ++nt)
        acc[mt][nt] = WMMA_BF16(a[mt], b[nt], acc[mt][nt]);
  }

  // Epilogue: c0/row0 are wave-uniform -> force SGPRs so the Q/K/V dispatch
  // is pure SALU + s_cbranch (no v_cmpx exec juggling, no VALU division).
#pragma unroll
  for (int mt = 0; mt < 2; ++mt) {
    const int row0  = __builtin_amdgcn_readfirstlane(m0 + mt * 16);
    const int b_    = row0 >> 10;
    const int sbase = (row0 & 1023) + hi8;
#pragma unroll
    for (int nt = 0; nt < 4; ++nt) {
      const int c0    = __builtin_amdgcn_readfirstlane(n0 + nt * 16);
      const int h     = c0 / 192;          // scalar division
      const int rem   = c0 - h * 192;
      const int which = rem >> 6;
      const int hd    = (rem & 63) + lo16;
      const int bh    = b_ * 16 + h;
      const float bias = bqkv[c0 + lo16];
      if (which == 0) {
        __bf16* qp = Qb + ((size_t)bh * 1024 + sbase) * 64 + hd;
#pragma unroll
        for (int r = 0; r < 8; ++r) qp[(size_t)r * 64] = (__bf16)(acc[mt][nt][r] + bias);
      } else if (which == 1) {
        __bf16* kp = Kb + ((size_t)bh * 1024 + sbase) * 64 + hd;
#pragma unroll
        for (int r = 0; r < 8; ++r) kp[(size_t)r * 64] = (__bf16)(acc[mt][nt][r] + bias);
      } else {
        __bf16* vp = Vt + ((size_t)bh * 64 + hd) * 1024 + sbase;
#pragma unroll
        for (int r = 0; r < 8; ++r) vp[r] = (__bf16)(acc[mt][nt][r] + bias);
      }
    }
  }
}

// ------------------------------ attention ----------------------------------
// Grid: B*H*(S/32) = 2048 blocks; 64 thr = 2 waves; wave owns 16 queries.
// LDS: 2 x 16 x 1028 fp32 = 131,584 B -> two blocks co-resident per WGP
// (263 KB <= 320 KB), i.e. 4 waves/WGP so all 4 SIMD32s stay busy.
#define SROW 1028
__global__ __launch_bounds__(64) void mha_attn(
    const __bf16* __restrict__ Qb,
    const __bf16* __restrict__ Kb,
    const __bf16* __restrict__ Vt,
    const float*  __restrict__ mask,   // [B,S,S]
    float*        __restrict__ att,    // [B*H,S,S]
    __bf16*       __restrict__ ctxb) { // [4096,1024] token-major
  extern __shared__ float smem[];
  const int wave = threadIdx.x >> 5, lane = threadIdx.x & 31;
  const int lo16 = lane & 15;
  const int hi8  = (lane >> 4) * 8;
  const int hi16 = (lane >> 4) * 16;
  const int bh = blockIdx.x >> 5, qblk = blockIdx.x & 31;
  const int b_ = bh >> 4, h = bh & 15;
  const int q0 = qblk * 32 + wave * 16;     // 16 queries per wave
  float* srow = smem + (size_t)wave * 16 * SROW;

  // Q fragments: 2 K-steps of 32 over hd=64
  v16bf aQ[2];
  {
    const __bf16* qp = Qb + ((size_t)bh * 1024 + q0 + lo16) * 64;
#pragma unroll
    for (int s = 0; s < 2; ++s) aQ[s] = fragA(qp + s * 32, hi8);
  }

  // scores = (Q K^T) * scale + mask  -> LDS (fp32)
  for (int j0 = 0; j0 < 1024; j0 += 16) {
    v8f acc = {};
    const __bf16* kp = Kb + ((size_t)bh * 1024 + j0 + lo16) * 64;
#pragma unroll
    for (int s = 0; s < 2; ++s) {
      v16bf bK = fragB(kp + s * 32, hi16);
      acc = WMMA_BF16(aQ[s], bK, acc);
    }
    const float* mrow = mask + ((size_t)b_ * 1024 + q0 + hi8) * 1024 + j0 + lo16;
#pragma unroll
    for (int r = 0; r < 8; ++r) {
      float sc = acc[r] * 0.125f + mrow[(size_t)r * 1024];
      srow[(hi8 + r) * SROW + j0 + lo16] = sc;
    }
  }

  // Per-row softmax (wave-local). Normalized P goes to att via nontemporal
  // b128 stores (268 MB stream, never re-read: keep it out of L2) and is
  // re-packed bf16 IN PLACE in the low half of the row (writes trail reads).
  for (int r = 0; r < 16; ++r) {
    float* row = srow + r * SROW;
    float m = -3.4e38f;
#pragma unroll 4
    for (int i = 0; i < 16; ++i) {
      v2f s2 = *(const v2f*)(row + 2 * lane + 64 * i);
      m = fmaxf(m, fmaxf(s2.x, s2.y));
    }
#pragma unroll
    for (int o = 16; o > 0; o >>= 1) m = fmaxf(m, __shfl_xor(m, o));
    float sum = 0.f;
#pragma unroll 4
    for (int i = 0; i < 16; ++i) {
      int c = 2 * lane + 64 * i;
      v2f s2 = *(const v2f*)(row + c);
      v2f e2;
      e2.x = __expf(s2.x - m);
      e2.y = __expf(s2.y - m);
      *(v2f*)(row + c) = e2;
      sum += e2.x + e2.y;
    }
#pragma unroll
    for (int o = 16; o > 0; o >>= 1) sum += __shfl_xor(sum, o);
    float inv = 1.0f / sum;
    float* arow = att + ((size_t)bh * 1024 + q0 + r) * 1024;
#pragma unroll 2
    for (int i = 0; i < 8; ++i) {
      int c = 4 * lane + 128 * i;
      v4f e4 = *(const v4f*)(row + c);
      v4f p4 = e4 * inv;
      __builtin_nontemporal_store(p4, (v4f*)(arow + c));      // NT b128
      uint2 pk;
      pk.x = pack_bf16x2(p4.x, p4.y);
      pk.y = pack_bf16x2(p4.z, p4.w);
      *(uint2*)((unsigned*)row + (c >> 1)) = pk;              // bf16 P row
    }
  }

  // ctx[16,64] = P[16,1024] @ V[1024,64] ; P A-fragments via ds_load_b128
  v8f cacc[4] = {};
  for (int j0 = 0; j0 < 1024; j0 += 32) {
    const __bf16* pb = (const __bf16*)(srow + lo16 * SROW);
    v16bf aP = fragA(pb + j0, hi8);
#pragma unroll
    for (int n = 0; n < 4; ++n) {
      const __bf16* vp = Vt + ((size_t)bh * 64 + n * 16 + lo16) * 1024 + j0;
      v16bf bV = fragB(vp, hi16);
      cacc[n] = WMMA_BF16(aP, bV, cacc[n]);
    }
  }

#pragma unroll
  for (int n = 0; n < 4; ++n) {
    __bf16* cp = ctxb + ((size_t)b_ * 1024 + q0 + hi8) * 1024 +
                 h * 64 + n * 16 + lo16;
#pragma unroll
    for (int r = 0; r < 8; ++r) cp[(size_t)r * 1024] = (__bf16)(cacc[n][r]);
  }
}

// ------------------------------ output GEMM --------------------------------
__global__ __launch_bounds__(256) void out_gemm(
    const __bf16* __restrict__ Cb,    // [4096,1024] bf16 ctx
    const __bf16* __restrict__ WoT,   // [1024,1024] bf16 (N-major)
    const float*  __restrict__ bo,    // [1024]
    float*        __restrict__ out) { // [4096,1024] fp32
  const int KDIM = 1024;
  const int wave = threadIdx.x >> 5, lane = threadIdx.x & 31;
  const int lo16 = lane & 15;
  const int hi8  = (lane >> 4) * 8;
  const int hi16 = (lane >> 4) * 16;
  const int m0 = blockIdx.y * 64  + (wave >> 2) * 32;
  const int n0 = blockIdx.x * 256 + (wave & 3) * 64;

  v8f acc[2][4] = {};
  for (int k0 = 0; k0 < KDIM; k0 += 32) {
    v16bf a[2], b[4];
#pragma unroll
    for (int mt = 0; mt < 2; ++mt)
      a[mt] = fragA(Cb + (size_t)(m0 + mt * 16 + lo16) * KDIM + k0, hi8);
#pragma unroll
    for (int nt = 0; nt < 4; ++nt)
      b[nt] = fragB(WoT + (size_t)(n0 + nt * 16 + lo16) * KDIM + k0, hi16);
#pragma unroll
    for (int mt = 0; mt < 2; ++mt)
#pragma unroll
      for (int nt = 0; nt < 4; ++nt)
        acc[mt][nt] = WMMA_BF16(a[mt], b[nt], acc[mt][nt]);
  }

#pragma unroll
  for (int mt = 0; mt < 2; ++mt)
#pragma unroll
    for (int nt = 0; nt < 4; ++nt) {
      const int c = n0 + nt * 16 + lo16;
      const float bias = bo[c];
      float* op = out + (size_t)(m0 + mt * 16 + hi8) * 1024 + c;
#pragma unroll
      for (int r = 0; r < 8; ++r)
        __builtin_nontemporal_store(acc[mt][nt][r] + bias, op + (size_t)r * 1024);
    }
}

// ------------------------------ launcher -----------------------------------
extern "C" void kernel_launch(void* const* d_in, const int* in_sizes, int n_in,
                              void* d_out, int out_size, void* d_ws, size_t ws_size,
                              hipStream_t stream) {
  const float* x    = (const float*)d_in[0];  // [4,1024,1024]
  const float* mask = (const float*)d_in[1];  // [4,1024,1024]
  const float* Wqkv = (const float*)d_in[2];  // [1024,3072]
  const float* bqkv = (const float*)d_in[3];  // [3072]
  const float* Wo   = (const float*)d_in[4];  // [1024,1024]
  const float* bo   = (const float*)d_in[5];  // [1024]

  float* out = (float*)d_out;                       // 4,194,304 fp32
  float* att = out + (size_t)4 * 1024 * 1024;       // 67,108,864 fp32

  char* ws = (char*)d_ws;
  auto carve = [&](size_t bytes) {
    char* p = ws;
    ws += (bytes + 255) & ~(size_t)255;
    return (void*)p;
  };
  const size_t TOK = 4096;
  __bf16* Xb    = (__bf16*)carve(TOK * 1024 * 2);
  __bf16* WqkvT = (__bf16*)carve((size_t)3072 * 1024 * 2);
  __bf16* WoT   = (__bf16*)carve((size_t)1024 * 1024 * 2);
  __bf16* Qb    = (__bf16*)carve((size_t)64 * 1024 * 64 * 2);
  __bf16* Kb    = (__bf16*)carve((size_t)64 * 1024 * 64 * 2);
  __bf16* Vt    = (__bf16*)carve((size_t)64 * 64 * 1024 * 2);
  __bf16* Ctx   = (__bf16*)carve(TOK * 1024 * 2);

  // 1) precision conversion / weight transposes
  cvt_f32_to_bf16_v4<<<(TOK * 1024 / 4 + 255) / 256, 256, 0, stream>>>(
      x, Xb, TOK * 1024 / 4);
  transpose_to_bf16<<<(1024 * 3072) / 256, 256, 0, stream>>>(Wqkv, WqkvT, 1024, 3072);
  transpose_to_bf16<<<(1024 * 1024) / 256, 256, 0, stream>>>(Wo, WoT, 1024, 1024);

  // 2) fused QKV projection -> per-head Q, K, V^T (bf16)
  qkv_gemm<<<dim3(3072 / 256, 4096 / 64), 256, 0, stream>>>(
      Xb, WqkvT, bqkv, Qb, Kb, Vt);

  // 3) attention: scores -> LDS softmax -> att (fp32 NT) + ctx (bf16)
  const size_t smemBytes = (size_t)2 * 16 * SROW * sizeof(float);  // 131,584 B
  mha_attn<<<dim3(4 * 16 * (1024 / 32)), 64, smemBytes, stream>>>(
      Qb, Kb, Vt, mask, att, Ctx);

  // 4) output projection
  out_gemm<<<dim3(1024 / 256, 4096 / 64), 256, 0, stream>>>(Ctx, WoT, bo, out);
}
